// GAT2bn_14336600834808
// MI455X (gfx1250) — compile-verified, matrix operand order checked by reading, not run on hardware
//
#include <hip/hip_runtime.h>
#include <cstddef>
#include <cstdint>

// ---------------------------------------------------------------------------
// GATv2 x2 + BN + pool + MLP head for MI455X (gfx1250, wave32, WMMA).
// Dense GEMMs -> v_wmma_f32_16x16x32_bf16; 128x64 block tile, 32x32/wave
// (4 WMMAs per wave per K-step), float4 staging loads, branchless M-clamp.
// Edge phase  -> 3 memory-lean passes with f32 atomics (segment softmax).
// ---------------------------------------------------------------------------

#define N_NODES 20000
#define N_EDGES 320000
#define N_GRAPH 128
#define EPS_BN  1e-5f
#define SLOPE   0.2f

typedef __attribute__((ext_vector_type(16))) __bf16 v16bf;
typedef __attribute__((ext_vector_type(8)))  float  v8f;

static __device__ __forceinline__ unsigned f2bf(float f) {
  unsigned int u = __float_as_uint(f);
  u += 0x7FFFu + ((u >> 16) & 1u);        // round-to-nearest-even
  return u >> 16;
}
static __device__ __forceinline__ unsigned pack2(float a, float b) {
  return f2bf(a) | (f2bf(b) << 16);
}
// order-preserving float<->uint mapping for atomicMax-based segment max
static __device__ __forceinline__ unsigned f2ord(float f) {
  unsigned u = __float_as_uint(f);
  return (u & 0x80000000u) ? ~u : (u | 0x80000000u);
}
static __device__ __forceinline__ float ord2f(unsigned u) {
  return __uint_as_float((u & 0x80000000u) ? (u & 0x7FFFFFFFu) : ~u);
}

union FragBF { v16bf v; unsigned int u[8]; };

static __device__ __forceinline__ void ld_frag(FragBF& f, const unsigned short* p) {
  uint4 lo = *(const uint4*)p;        // K/N sub-block elements 0..7
  uint4 hi = *(const uint4*)(p + 16); // elements 8..15 (K+16)
  f.u[0] = lo.x; f.u[1] = lo.y; f.u[2] = lo.z; f.u[3] = lo.w;
  f.u[4] = hi.x; f.u[5] = hi.y; f.u[6] = hi.z; f.u[7] = hi.w;
}

// ---------------------------------------------------------------------------
// Tiled WMMA GEMM: C[M,Nc] = A[M,K] @ B[K,Nc] (+bias) (+relu), fp32 in/out,
// bf16 multiply / f32 accumulate. Block = 256 thr (8 waves), tile 128x64,
// wave tile 32x32 (4 WMMAs / K-step). Requires K%32==0, Nc%64==0.
// ---------------------------------------------------------------------------
__global__ __launch_bounds__(256) void gemm_bf16_wmma(
    const float* __restrict__ A, const float* __restrict__ B,
    const float* __restrict__ bias, float* __restrict__ C,
    int M, int K, int Nc, int do_relu)
{
  __shared__ __align__(16) unsigned short As[128][40];  // A tile, row-major
  __shared__ __align__(16) unsigned short Bs[64][40];   // B tile, transposed: Bs[n][k]
  const int n0   = blockIdx.x * 64;
  const int m0   = blockIdx.y * 128;
  const int tid  = threadIdx.x;
  const int wave = tid >> 5, lane = tid & 31;
  const int wy   = wave >> 1, wx = wave & 1;            // 4x2 wave grid
  const int lm   = lane & 15;
  const int ko   = (lane >> 4) * 8;                     // K sub-block per lane half

  v8f acc00 = {}, acc01 = {}, acc10 = {}, acc11 = {};

  for (int kk = 0; kk < K; kk += 32) {
    __syncthreads();
    // A tile (128x32): float4 loads from clamped row, branchless zero-fill,
    // packed bf16 x4 stored as 8B. Coalesced along K.
    #pragma unroll
    for (int i = 0; i < 4; ++i) {
      int idx = tid + (i << 8);            // 0..1023
      int r = idx >> 3, c4 = (idx & 7) << 2;
      int gm  = m0 + r;
      int gmc = (gm < M) ? gm : (M - 1);
      float4 v = *(const float4*)(A + (size_t)gmc * K + kk + c4);
      if (gm >= M) { v.x = 0.f; v.y = 0.f; v.z = 0.f; v.w = 0.f; }
      uint2 p; p.x = pack2(v.x, v.y); p.y = pack2(v.z, v.w);
      *(uint2*)&As[r][c4] = p;
    }
    // B tile (32x64): float4 loads coalesced along N, transposed b16 scatter.
    #pragma unroll
    for (int i = 0; i < 2; ++i) {
      int idx = tid + (i << 8);            // 0..511
      int k = idx >> 4, n4 = (idx & 15) << 2;
      float4 v = *(const float4*)(B + (size_t)(kk + k) * Nc + n0 + n4);
      Bs[n4 + 0][k] = (unsigned short)f2bf(v.x);
      Bs[n4 + 1][k] = (unsigned short)f2bf(v.y);
      Bs[n4 + 2][k] = (unsigned short)f2bf(v.z);
      Bs[n4 + 3][k] = (unsigned short)f2bf(v.w);
    }
    __syncthreads();

    FragBF fa0, fa1, fb0, fb1;
    ld_frag(fa0, &As[wy * 32 +      lm][ko]);
    ld_frag(fa1, &As[wy * 32 + 16 + lm][ko]);
    ld_frag(fb0, &Bs[wx * 32 +      lm][ko]);
    ld_frag(fb1, &Bs[wx * 32 + 16 + lm][ko]);

    acc00 = __builtin_amdgcn_wmma_f32_16x16x32_bf16(false, fa0.v, false, fb0.v,
                                                    (short)0, acc00, false, false);
    acc01 = __builtin_amdgcn_wmma_f32_16x16x32_bf16(false, fa0.v, false, fb1.v,
                                                    (short)0, acc01, false, false);
    acc10 = __builtin_amdgcn_wmma_f32_16x16x32_bf16(false, fa1.v, false, fb0.v,
                                                    (short)0, acc10, false, false);
    acc11 = __builtin_amdgcn_wmma_f32_16x16x32_bf16(false, fa1.v, false, fb1.v,
                                                    (short)0, acc11, false, false);
  }

  // epilogue: D layout -> lane l: col = l&15, VGPR r: row = (l>>4)*8 + r
  const int r0  = (lane >> 4) * 8;
  const int cn0 = n0 + wx * 32 + lm;
  #pragma unroll
  for (int r = 0; r < 8; ++r) {
    int gm0 = m0 + wy * 32 + r0 + r;       // rows for acc0x
    int gm1 = gm0 + 16;                    // rows for acc1x
    float b0 = bias ? bias[cn0]      : 0.f;
    float b1 = bias ? bias[cn0 + 16] : 0.f;
    if (gm0 < M) {
      float v0 = acc00[r] + b0, v1 = acc01[r] + b1;
      if (do_relu) { v0 = fmaxf(v0, 0.f); v1 = fmaxf(v1, 0.f); }
      C[(size_t)gm0 * Nc + cn0]      = v0;
      C[(size_t)gm0 * Nc + cn0 + 16] = v1;
    }
    if (gm1 < M) {
      float v0 = acc10[r] + b0, v1 = acc11[r] + b1;
      if (do_relu) { v0 = fmaxf(v0, 0.f); v1 = fmaxf(v1, 0.f); }
      C[(size_t)gm1 * Nc + cn0]      = v0;
      C[(size_t)gm1 * Nc + cn0 + 16] = v1;
    }
  }
}

// ---------------------------------------------------------------------------
// Edge pass 1: score[e][h] = sum_c leaky(xl[src]+xr[dst]+ee)[h,c] * att[h,c]
// + atomic segment-max over dst. One wave per edge; we/att staged in LDS;
// ee computed on the fly (K=8) to avoid a [E,HC] intermediate.
// ---------------------------------------------------------------------------
__global__ __launch_bounds__(256) void edge_score_kernel(
    const float* __restrict__ XL, const float* __restrict__ XR,
    const float* __restrict__ ea, const float* __restrict__ we,
    const float* __restrict__ att, const int* __restrict__ src,
    const int* __restrict__ dst, float* __restrict__ score,
    unsigned* __restrict__ smax, int E, int HC, int C)
{
  __shared__ float s_we[8 * 1024];
  __shared__ float s_att[1024];
  for (int i = threadIdx.x; i < 8 * HC; i += 256) s_we[i] = we[i];
  for (int i = threadIdx.x; i < HC; i += 256)     s_att[i] = att[i];
  __syncthreads();

  const int wave = threadIdx.x >> 5, lane = threadIdx.x & 31;
  const int jmax = C >> 5;                       // 32-wide chunks per head
  for (int e = blockIdx.x * 8 + wave; e < E; e += gridDim.x * 8) {
    const int s = src[e], d = dst[e];
    float eav = (lane < 8) ? ea[(size_t)e * 8 + lane] : 0.f;
    float eak[8];
    #pragma unroll
    for (int k = 0; k < 8; ++k) eak[k] = __shfl(eav, k, 32);

    const float* xlp = XL + (size_t)s * HC;
    const float* xrp = XR + (size_t)d * HC;
    #pragma unroll
    for (int h = 0; h < 4; ++h) {
      float r = 0.f;
      for (int j = 0; j < jmax; ++j) {
        int c = h * C + (j << 5) + lane;
        float ee = 0.f;
        #pragma unroll
        for (int k = 0; k < 8; ++k) ee += eak[k] * s_we[k * HC + c];
        float m = xlp[c] + xrp[c] + ee;
        float t = (m > 0.f) ? m : SLOPE * m;
        r += t * s_att[c];
      }
      #pragma unroll
      for (int off = 16; off > 0; off >>= 1) r += __shfl_xor(r, off, 32);
      if (lane == h) {
        score[(size_t)e * 4 + h] = r;
        atomicMax(&smax[d * 4 + h], f2ord(r));
      }
    }
  }
}

// Edge pass 2: ex = exp(score - smax[dst]); denom[dst] += ex (in-place ex).
__global__ void edge_exp_kernel(float* __restrict__ score,
                                const unsigned* __restrict__ smax,
                                const int* __restrict__ dst,
                                float* __restrict__ denom, int total)
{
  int tid = blockIdx.x * blockDim.x + threadIdx.x;
  if (tid >= total) return;
  int e = tid >> 2, h = tid & 3;
  int d = dst[e];
  float mx = ord2f(smax[d * 4 + h]);
  float ex = __expf(score[tid] - mx);
  score[tid] = ex;
  atomicAdd(&denom[d * 4 + h], ex);
}

// Edge pass 3: out[dst] += (ex/denom[dst])[h] * xl[src][h,c]. Wave per edge.
__global__ __launch_bounds__(256) void edge_scatter_kernel(
    const float* __restrict__ XL, const float* __restrict__ ex,
    const float* __restrict__ denom, const int* __restrict__ src,
    const int* __restrict__ dst, float* __restrict__ out,
    int E, int HC, int C)
{
  const int wave = threadIdx.x >> 5, lane = threadIdx.x & 31;
  const int jmax = C >> 5;
  for (int e = blockIdx.x * 8 + wave; e < E; e += gridDim.x * 8) {
    const int s = src[e], d = dst[e];
    float al = 0.f;
    if (lane < 4) al = ex[(size_t)e * 4 + lane] / denom[d * 4 + lane];
    float av0 = __shfl(al, 0, 32), av1 = __shfl(al, 1, 32);
    float av2 = __shfl(al, 2, 32), av3 = __shfl(al, 3, 32);
    float av[4] = {av0, av1, av2, av3};
    const float* xlp = XL + (size_t)s * HC;
    float* op = out + (size_t)d * HC;
    #pragma unroll
    for (int h = 0; h < 4; ++h) {
      float a = av[h];
      for (int j = 0; j < jmax; ++j) {
        int c = h * C + (j << 5) + lane;
        atomicAdd(&op[c], a * xlp[c]);
      }
    }
  }
}

// Elementwise: x = relu(x + bias[c])
__global__ void bias_relu_kernel(float* __restrict__ X,
                                 const float* __restrict__ bias,
                                 int total, int Cn)
{
  int tid = blockIdx.x * blockDim.x + threadIdx.x;
  if (tid >= total) return;
  float v = X[tid] + bias[tid & (Cn - 1)];
  X[tid] = (v > 0.f) ? v : 0.f;
}

// BN stats: per-channel sum / sumsq via per-block partials + atomics.
__global__ void bn_stats_kernel(const float* __restrict__ X,
                                float* __restrict__ sums, int Nn, int Cn)
{
  int c = blockIdx.y * 256 + threadIdx.x;
  float s = 0.f, q = 0.f;
  for (int n = blockIdx.x; n < Nn; n += gridDim.x) {
    float v = X[(size_t)n * Cn + c];
    s += v; q += v * v;
  }
  atomicAdd(&sums[c], s);
  atomicAdd(&sums[Cn + c], q);
}

__global__ void bn_finalize_kernel(float* __restrict__ sums, int Nn, int Cn)
{
  int c = blockIdx.x * blockDim.x + threadIdx.x;
  if (c >= Cn) return;
  float inv_n = 1.f / (float)Nn;
  float mean = sums[c] * inv_n;
  float var  = sums[Cn + c] * inv_n - mean * mean;
  sums[c] = mean;
  sums[Cn + c] = rsqrtf(var + EPS_BN);
}

__global__ void bn_apply_kernel(float* __restrict__ X,
                                const float* __restrict__ st,
                                const float* __restrict__ g,
                                const float* __restrict__ b,
                                int total, int Cn)
{
  int tid = blockIdx.x * blockDim.x + threadIdx.x;
  if (tid >= total) return;
  int c = tid & (Cn - 1);
  X[tid] = (X[tid] - st[c]) * st[Cn + c] * g[c] + b[c];
}

// global_add_pool: pooled[batch[n]][c] += X[n][c]
__global__ void pool_kernel(const float* __restrict__ X,
                            const int* __restrict__ batch,
                            float* __restrict__ pooled, int Nn)
{
  int tid = blockIdx.x * blockDim.x + threadIdx.x;
  if (tid >= Nn * 256) return;
  int n = tid >> 8, c = tid & 255;
  atomicAdd(&pooled[batch[n] * 256 + c], X[tid]);
}

// head: out[g] = relu(pool[g]@fc1 + b1) @ fc2 + b2. One block per graph.
__global__ __launch_bounds__(64) void head_kernel(
    const float* __restrict__ pooled, const float* __restrict__ fc1_w,
    const float* __restrict__ fc1_b, const float* __restrict__ fc2_w,
    const float* __restrict__ fc2_b, float* __restrict__ out)
{
  __shared__ float sh[64];
  int gi = blockIdx.x, j = threadIdx.x;
  const float* grow = pooled + (size_t)gi * 256;
  float s = fc1_b[j];
  for (int k = 0; k < 256; ++k) s += grow[k] * fc1_w[k * 64 + j];
  s = fmaxf(s, 0.f);
  sh[j] = s * fc2_w[j];
  __syncthreads();
  if (j == 0) {
    float r = fc2_b[0];
    #pragma unroll
    for (int k = 0; k < 64; ++k) r += sh[k];
    out[gi] = r;
  }
}

// ---------------------------------------------------------------------------
extern "C" void kernel_launch(void* const* d_in, const int* in_sizes, int n_in,
                              void* d_out, int out_size, void* d_ws, size_t ws_size,
                              hipStream_t stream) {
  (void)in_sizes; (void)n_in; (void)out_size; (void)ws_size;
  const float* x        = (const float*)d_in[0];
  const float* edge_attr= (const float*)d_in[1];
  const int*   ei       = (const int*)  d_in[2];
  const int*   batch    = (const int*)  d_in[3];
  const float* dr1_w = (const float*)d_in[4];  const float* dr1_b = (const float*)d_in[5];
  const float* dr2_w = (const float*)d_in[6];  const float* dr2_b = (const float*)d_in[7];
  const float* c1_wl = (const float*)d_in[8];  const float* c1_wr = (const float*)d_in[9];
  const float* c1_we = (const float*)d_in[10]; const float* c1_att= (const float*)d_in[11];
  const float* c1_b  = (const float*)d_in[12];
  const float* bn1_g = (const float*)d_in[13]; const float* bn1_b = (const float*)d_in[14];
  const float* c2_wl = (const float*)d_in[15]; const float* c2_wr = (const float*)d_in[16];
  const float* c2_we = (const float*)d_in[17]; const float* c2_att= (const float*)d_in[18];
  const float* c2_b  = (const float*)d_in[19];
  const float* bn2_g = (const float*)d_in[20]; const float* bn2_b = (const float*)d_in[21];
  const float* fc1_w = (const float*)d_in[22]; const float* fc1_b = (const float*)d_in[23];
  const float* fc2_w = (const float*)d_in[24]; const float* fc2_b = (const float*)d_in[25];
  float* out = (float*)d_out;

  const int* src = ei;
  const int* dst = ei + N_EDGES;

  // ---- workspace layout (floats), ~216 MB total ----
  float* ws = (float*)d_ws;
  float* H1   = ws;                                   // [N,256]
  float* H2   = H1  + (size_t)N_NODES * 256;          // [N,64]
  float* XL1  = H2  + (size_t)N_NODES * 64;           // [N,1024]
  float* XR1  = XL1 + (size_t)N_NODES * 1024;         // [N,1024]
  float* SC   = XR1 + (size_t)N_NODES * 1024;         // [E,4]
  unsigned* SMAX = (unsigned*)(SC + (size_t)N_EDGES * 4);   // [N,4]
  float* DEN  = (float*)(SMAX + (size_t)N_NODES * 4); // [N,4]
  float* STATS= DEN + (size_t)N_NODES * 4;            // [2*1024]
  float* OUT2 = STATS + 2048;                         // [N,256]
  float* POOL = OUT2 + (size_t)N_NODES * 256;         // [G,256]
  // aliases (sequential reuse)
  float* OUT1 = XR1;   // conv1 output / BN1'd features (XR1 dead after pass1)
  float* XL2  = XL1;   // conv2 xl (XL1 dead after conv1 scatter)
  float* XR2  = H1;    // conv2 xr (H1 dead after GEMM2)

  const int MB = (N_NODES + 127) / 128;  // 157 M-tiles
  dim3 thr(256);

  // ---- encoder GEMMs ----
  gemm_bf16_wmma<<<dim3(256/64, MB), thr, 0, stream>>>(x,  dr1_w, dr1_b, H1, N_NODES, 512, 256, 1);
  gemm_bf16_wmma<<<dim3(64/64,  MB), thr, 0, stream>>>(H1, dr2_w, dr2_b, H2, N_NODES, 256, 64, 0);

  // ---- conv1 projections ----
  gemm_bf16_wmma<<<dim3(1024/64, MB), thr, 0, stream>>>(H2, c1_wl, nullptr, XL1, N_NODES, 64, 1024, 0);
  gemm_bf16_wmma<<<dim3(1024/64, MB), thr, 0, stream>>>(H2, c1_wr, nullptr, XR1, N_NODES, 64, 1024, 0);

  // ---- conv1 edge phase ----
  hipMemsetAsync(SMAX, 0, (size_t)N_NODES * 4 * sizeof(unsigned), stream);
  hipMemsetAsync(DEN,  0, (size_t)N_NODES * 4 * sizeof(float), stream);
  edge_score_kernel<<<2048, thr, 0, stream>>>(XL1, XR1, edge_attr, c1_we, c1_att,
                                              src, dst, SC, SMAX, N_EDGES, 1024, 256);
  hipMemsetAsync(OUT1, 0, (size_t)N_NODES * 1024 * sizeof(float), stream);
  edge_exp_kernel<<<(N_EDGES * 4 + 255) / 256, thr, 0, stream>>>(SC, SMAX, dst, DEN, N_EDGES * 4);
  edge_scatter_kernel<<<2048, thr, 0, stream>>>(XL1, SC, DEN, src, dst, OUT1, N_EDGES, 1024, 256);

  // ---- bias + relu + BN1 (in place on OUT1) ----
  bias_relu_kernel<<<((size_t)N_NODES * 1024 + 255) / 256, thr, 0, stream>>>(OUT1, c1_b, N_NODES * 1024, 1024);
  hipMemsetAsync(STATS, 0, 2 * 1024 * sizeof(float), stream);
  bn_stats_kernel<<<dim3(64, 1024 / 256), thr, 0, stream>>>(OUT1, STATS, N_NODES, 1024);
  bn_finalize_kernel<<<(1024 + 255) / 256, thr, 0, stream>>>(STATS, N_NODES, 1024);
  bn_apply_kernel<<<((size_t)N_NODES * 1024 + 255) / 256, thr, 0, stream>>>(OUT1, STATS, bn1_g, bn1_b, N_NODES * 1024, 1024);

  // ---- conv2 projections (input = BN1 features in OUT1) ----
  gemm_bf16_wmma<<<dim3(256/64, MB), thr, 0, stream>>>(OUT1, c2_wl, nullptr, XL2, N_NODES, 1024, 256, 0);
  gemm_bf16_wmma<<<dim3(256/64, MB), thr, 0, stream>>>(OUT1, c2_wr, nullptr, XR2, N_NODES, 1024, 256, 0);

  // ---- conv2 edge phase ----
  hipMemsetAsync(SMAX, 0, (size_t)N_NODES * 4 * sizeof(unsigned), stream);
  hipMemsetAsync(DEN,  0, (size_t)N_NODES * 4 * sizeof(float), stream);
  edge_score_kernel<<<2048, thr, 0, stream>>>(XL2, XR2, edge_attr, c2_we, c2_att,
                                              src, dst, SC, SMAX, N_EDGES, 256, 64);
  hipMemsetAsync(OUT2, 0, (size_t)N_NODES * 256 * sizeof(float), stream);
  edge_exp_kernel<<<(N_EDGES * 4 + 255) / 256, thr, 0, stream>>>(SC, SMAX, dst, DEN, N_EDGES * 4);
  edge_scatter_kernel<<<2048, thr, 0, stream>>>(XL2, SC, DEN, src, dst, OUT2, N_EDGES, 256, 64);

  // ---- bias + relu + BN2 ----
  bias_relu_kernel<<<((size_t)N_NODES * 256 + 255) / 256, thr, 0, stream>>>(OUT2, c2_b, N_NODES * 256, 256);
  hipMemsetAsync(STATS, 0, 2 * 256 * sizeof(float), stream);
  bn_stats_kernel<<<dim3(64, 256 / 256), thr, 0, stream>>>(OUT2, STATS, N_NODES, 256);
  bn_finalize_kernel<<<1, thr, 0, stream>>>(STATS, N_NODES, 256);
  bn_apply_kernel<<<((size_t)N_NODES * 256 + 255) / 256, thr, 0, stream>>>(OUT2, STATS, bn2_g, bn2_b, N_NODES * 256, 256);

  // ---- pool + head ----
  hipMemsetAsync(POOL, 0, (size_t)N_GRAPH * 256 * sizeof(float), stream);
  pool_kernel<<<((size_t)N_NODES * 256 + 255) / 256, thr, 0, stream>>>(OUT2, batch, POOL, N_NODES);
  head_kernel<<<N_GRAPH, 64, 0, stream>>>(POOL, fc1_w, fc1_b, fc2_w, fc2_b, out);
}
